// MultiHeadSelfAttention_40690520162798
// MI455X (gfx1250) — matrix operand order repaired and checked by reading, not compiled, not run on hardware
//
#include <hip/hip_runtime.h>
#include <math.h>

#define EMB   1024
#define HEADS 16
#define HD    64
#define BATCH 4
#define SEQ   2048
#define TOK   (BATCH * SEQ)   // 8192 tokens

typedef __attribute__((ext_vector_type(16))) __bf16 v16bf;
typedef __attribute__((ext_vector_type(8)))  float  v8f;

__device__ inline v8f wmma_bf16(v16bf a, v16bf b, v8f c) {
  return __builtin_amdgcn_wmma_f32_16x16x32_bf16(false, a, false, b, (short)0, c,
                                                 false, false);
}

// ---- Fragment loaders (CDNA5 wave32 WMMA 16x16x32 bf16 layouts) -------------
// A (16x32, MxK): lane holds row m = lane&15.
//   VGPR j (j<4):  K = 8*half + 2j, +1 ;  VGPR j (>=4): K = 16 + 8*half + 2(j-4), +1
__device__ inline v16bf load_a_frag(const __bf16* p, int ld, int lane) {
  const int m = lane & 15, hf = lane >> 4;
  const __bf16* row = p + (size_t)m * ld;
  union { v16bf v; unsigned u[8]; } fa;
#pragma unroll
  for (int j = 0; j < 4; ++j) {
    fa.u[j]     = *(const unsigned*)(row + 8 * hf + 2 * j);
    fa.u[4 + j] = *(const unsigned*)(row + 16 + 8 * hf + 2 * j);
  }
  return fa.v;
}

// B (32x16, KxN) where B[k][n] is stored k-contiguously per n:
//   lane holds col n = lane&15; VGPR j (j<4): K = 8*half + 2j, +1 ; VGPR j (>=4): K = 16 + 8*half + 2(j-4), +1
__device__ inline v16bf load_b_frag_rows(const __bf16* p, int ld, int lane) {
  const int n = lane & 15, hf = lane >> 4;
  const __bf16* row = p + (size_t)n * ld + 8 * hf;
  union { v16bf v; unsigned u[8]; } fb;
#pragma unroll
  for (int j = 0; j < 4; ++j) {
    fb.u[j]     = *(const unsigned*)(row + 2 * j);
    fb.u[4 + j] = *(const unsigned*)(row + 16 + 2 * j);
  }
  return fb.v;
}

// ---- fp32 -> bf16 conversion ------------------------------------------------
__global__ void cvt_bf16(const float* __restrict__ in, __bf16* __restrict__ out, int n) {
  int i = blockIdx.x * blockDim.x + threadIdx.x;
  int stride = gridDim.x * blockDim.x;
  for (; i < n; i += stride) out[i] = (__bf16)in[i];
}

// ---- QKV projection: qkv = x @ Wqkv^T + b -----------------------------------
// Scatter: Q,K -> [which][b][h][s][d]; V -> transposed [b][h][d][s] so the
// attention P@V B-fragments become contiguous row loads.
// One wave computes a 32(M) x 64(N) tile. 256 m-tiles x 48 n-tiles = 12288 waves.
__global__ __launch_bounds__(128) void qkv_gemm(const __bf16* __restrict__ X,
                                                const __bf16* __restrict__ W,
                                                const float* __restrict__ bias,
                                                __bf16* __restrict__ QKV) {
  const int lane = threadIdx.x & 31;
  const int wid  = blockIdx.x * 4 + (threadIdx.x >> 5);
  const int mt = wid & 255, nt = wid >> 8;
  const int m0 = mt * 32, n0 = nt * 64;
  const v8f z = {0.f, 0.f, 0.f, 0.f, 0.f, 0.f, 0.f, 0.f};
  v8f c[2][4];
#pragma unroll
  for (int mi = 0; mi < 2; ++mi)
#pragma unroll
    for (int t = 0; t < 4; ++t) c[mi][t] = z;

  const __bf16* arow0 = X + (size_t)(m0 + (lane & 15)) * EMB;
  const __bf16* arow1 = X + (size_t)(m0 + 16 + (lane & 15)) * EMB;

  for (int k0 = 0; k0 < EMB; k0 += 32) {
    if (k0 + 128 < EMB) {   // stream X (HBM-resident) ahead into caches
      __builtin_prefetch(arow0 + k0 + 128, 0, 1);
      __builtin_prefetch(arow1 + k0 + 128, 0, 1);
    }
    v16bf a0 = load_a_frag(X + (size_t)m0 * EMB + k0, EMB, lane);
    v16bf a1 = load_a_frag(X + (size_t)(m0 + 16) * EMB + k0, EMB, lane);
#pragma unroll
    for (int t = 0; t < 4; ++t) {
      v16bf b = load_b_frag_rows(W + (size_t)(n0 + 16 * t) * EMB + k0, EMB, lane);
      c[0][t] = wmma_bf16(a0, b, c[0][t]);
      c[1][t] = wmma_bf16(a1, b, c[1][t]);
    }
  }
  const int hf = lane >> 4, nl = lane & 15;
  const size_t tensor = (size_t)BATCH * HEADS * SEQ * HD;
#pragma unroll
  for (int mi = 0; mi < 2; ++mi)
#pragma unroll
    for (int t = 0; t < 4; ++t)
#pragma unroll
      for (int r = 0; r < 8; ++r) {
        const int m = m0 + mi * 16 + r + 8 * hf;       // token index
        const int n = n0 + t * 16 + nl;                // col in [0, 3072)
        const float v = c[mi][t][r] + bias[n];
        const int which = n >> 10, rem = n & 1023;
        const int hh = rem >> 6, dd = rem & 63;
        const int bb = m >> 11, ss = m & 2047;
        size_t dst;
        if (which == 2)  // V transposed: [b][h][d][s]
          dst = 2 * tensor + ((size_t)(bb * HEADS + hh) * HD + dd) * SEQ + ss;
        else             // Q, K: [which][b][h][s][d]
          dst = (size_t)which * tensor +
                ((size_t)(bb * HEADS + hh) * SEQ + ss) * HD + dd;
        QKV[dst] = (__bf16)v;
      }
}

// ---- online softmax update for one 16-query subtile -------------------------
// st0/st1: S^T score tiles (rows = keys kk+r+8hf / +16, col = q = lane&15),
// already scaled and causal-masked. Updates (m_q,l_q), rescales o[0..3]
// (rows q = r+8hf), returns P as a ready A-fragment (S^T C-layout maps
// element-for-element onto the A-fragment layout: zero cross-lane traffic).
__device__ inline v16bf softmax_update(v8f st0, v8f st1, float& m_q, float& l_q,
                                       v8f o[4], int hf) {
  float tmax = -INFINITY;
#pragma unroll
  for (int r = 0; r < 8; ++r) tmax = fmaxf(tmax, fmaxf(st0[r], st1[r]));
  tmax = fmaxf(tmax, __shfl_xor(tmax, 16, 32));
  const float mnew = fmaxf(m_q, tmax);
  float tsum = 0.f;
#pragma unroll
  for (int r = 0; r < 8; ++r) {
    st0[r] = __expf(st0[r] - mnew);
    st1[r] = __expf(st1[r] - mnew);
    tsum += st0[r] + st1[r];
  }
  tsum += __shfl_xor(tsum, 16, 32);
  const float alpha = __expf(m_q - mnew);
  l_q = l_q * alpha + tsum;
  m_q = mnew;
#pragma unroll
  for (int r = 0; r < 8; ++r) {
    const float ar = __shfl(alpha, r + 8 * hf, 32);
#pragma unroll
    for (int t = 0; t < 4; ++t) o[t][r] *= ar;
  }
  v16bf pa;
#pragma unroll
  for (int e = 0; e < 8; ++e) {
    pa[e]     = (__bf16)st0[e];
    pa[8 + e] = (__bf16)st1[e];
  }
  return pa;
}

// ---- Flash attention: one wave per (b, h, 32-query tile) --------------------
// Two 16-query subtiles share every K and V fragment (2x arithmetic intensity).
// Computes S^T = K @ Q^T so softmax row-reductions are a single shfl_xor(16).
__global__ __launch_bounds__(128) void attn_kernel(const __bf16* __restrict__ QKV,
                                                   __bf16* __restrict__ AO) {
  const int lane = threadIdx.x & 31;
  const int wid  = blockIdx.x * 4 + (threadIdx.x >> 5);   // 4096 waves
  const int qt = wid & 63;
  const int bh = wid >> 6;
  const int h = bh & 15, b = bh >> 4;
  const int q0 = qt * 32;                                 // first query of tile
  const size_t plane  = (size_t)SEQ * HD;
  const size_t tensor = (size_t)BATCH * HEADS * plane;
  const __bf16* Qp  = QKV + (size_t)(b * HEADS + h) * plane;            // [s][d]
  const __bf16* Kp  = Qp + tensor;                                      // [s][d]
  const __bf16* VTp = QKV + 2 * tensor + (size_t)(b * HEADS + h) * plane; // [d][s]
  const int hf = lane >> 4, nl = lane & 15;
  const v8f z = {0.f, 0.f, 0.f, 0.f, 0.f, 0.f, 0.f, 0.f};

  // Q^T B-fragments, both subtiles, both 32-dim halves (reused for all keys)
  const v16bf bq0a = load_b_frag_rows(Qp + (size_t)q0 * HD + 0,  HD, lane);
  const v16bf bq1a = load_b_frag_rows(Qp + (size_t)q0 * HD + 32, HD, lane);
  const v16bf bq0b = load_b_frag_rows(Qp + (size_t)(q0 + 16) * HD + 0,  HD, lane);
  const v16bf bq1b = load_b_frag_rows(Qp + (size_t)(q0 + 16) * HD + 32, HD, lane);

  v8f oA[4], oB[4];
#pragma unroll
  for (int t = 0; t < 4; ++t) { oA[t] = z; oB[t] = z; }
  float mA = -INFINITY, lA = 0.f;   // stats for q = q0 + nl       (lane-pair replicated)
  float mB = -INFINITY, lB = 0.f;   // stats for q = q0 + 16 + nl
  const float scale = 0.125f;       // 1/sqrt(64)
  const int qgA = q0 + nl, qgB = q0 + 16 + nl;

  for (int kk = 0; kk < q0 + 32; kk += 32) {
    if (kk + 32 < q0 + 32) {  // prefetch next key tile (K rows + VT row segments)
      __builtin_prefetch(Kp + (size_t)(kk + 32 + lane) * HD, 0, 1);
      __builtin_prefetch(VTp + (size_t)lane * SEQ + kk + 32, 0, 1);
      __builtin_prefetch(VTp + (size_t)(lane + 32) * SEQ + kk + 32, 0, 1);
    }
    // S^T tiles: rows = keys, cols = queries. Each K fragment feeds 2 WMMAs.
    v16bf ka = load_a_frag(Kp + (size_t)kk * HD, HD, lane);
    v8f st0a = wmma_bf16(ka, bq0a, z);
    v8f st0b = wmma_bf16(ka, bq0b, z);
    ka = load_a_frag(Kp + (size_t)kk * HD + 32, HD, lane);
    st0a = wmma_bf16(ka, bq1a, st0a);
    st0b = wmma_bf16(ka, bq1b, st0b);
    ka = load_a_frag(Kp + (size_t)(kk + 16) * HD, HD, lane);
    v8f st1a = wmma_bf16(ka, bq0a, z);
    v8f st1b = wmma_bf16(ka, bq0b, z);
    ka = load_a_frag(Kp + (size_t)(kk + 16) * HD + 32, HD, lane);
    st1a = wmma_bf16(ka, bq1a, st1a);
    st1b = wmma_bf16(ka, bq1b, st1b);

#pragma unroll
    for (int r = 0; r < 8; ++r) {
      st0a[r] *= scale; st1a[r] *= scale;
      st0b[r] *= scale; st1b[r] *= scale;
    }
    if (kk + 31 > q0) {  // diagonal-straddling region: exact causal mask
#pragma unroll
      for (int r = 0; r < 8; ++r) {
        const int key0 = kk + r + 8 * hf;
        if (key0 > qgA)      st0a[r] = -INFINITY;
        if (key0 + 16 > qgA) st1a[r] = -INFINITY;
        if (key0 > qgB)      st0b[r] = -INFINITY;
        if (key0 + 16 > qgB) st1b[r] = -INFINITY;
      }
    }
    const v16bf paA = softmax_update(st0a, st1a, mA, lA, oA, hf);
    const v16bf paB = softmax_update(st0b, st1b, mB, lB, oB, hf);

    // O += P @ V ; V^T rows are contiguous -> b128-coalesced B fragments,
    // each shared by both query subtiles.
#pragma unroll
    for (int t = 0; t < 4; ++t) {
      v16bf vb = load_b_frag_rows(VTp + (size_t)(16 * t) * SEQ + kk, SEQ, lane);
      oA[t] = wmma_bf16(paA, vb, oA[t]);
      oB[t] = wmma_bf16(paB, vb, oB[t]);
    }
  }
  // normalize and store bf16 attention output in [b][s][e] layout
#pragma unroll
  for (int r = 0; r < 8; ++r) {
    const float invA = 1.0f / __shfl(lA, r + 8 * hf, 32);
    const float invB = 1.0f / __shfl(lB, r + 8 * hf, 32);
    const int tokA = b * SEQ + q0 + r + 8 * hf;
    const size_t baseA = (size_t)tokA * EMB + h * HD;
    const size_t baseB = baseA + (size_t)16 * EMB;
#pragma unroll
    for (int t = 0; t < 4; ++t) {
      AO[baseA + 16 * t + nl] = (__bf16)(oA[t][r] * invA);
      AO[baseB + 16 * t + nl] = (__bf16)(oB[t][r] * invB);
    }
  }
}

// ---- Output projection: out = attn @ Wout^T + b_out (fp32 store) ------------
__global__ __launch_bounds__(128) void out_gemm(const __bf16* __restrict__ A,
                                                const __bf16* __restrict__ W,
                                                const float* __restrict__ bias,
                                                float* __restrict__ OUT) {
  const int lane = threadIdx.x & 31;
  const int wid  = blockIdx.x * 4 + (threadIdx.x >> 5);   // 4096 waves
  const int mt = wid & 255, nt = wid >> 8;                // 256 x 16 tiles
  const int m0 = mt * 32, n0 = nt * 64;
  const v8f z = {0.f, 0.f, 0.f, 0.f, 0.f, 0.f, 0.f, 0.f};
  v8f c[2][4];
#pragma unroll
  for (int mi = 0; mi < 2; ++mi)
#pragma unroll
    for (int t = 0; t < 4; ++t) c[mi][t] = z;

  const __bf16* arow0 = A + (size_t)(m0 + (lane & 15)) * EMB;
  const __bf16* arow1 = A + (size_t)(m0 + 16 + (lane & 15)) * EMB;

  for (int k0 = 0; k0 < EMB; k0 += 32) {
    if (k0 + 128 < EMB) {
      __builtin_prefetch(arow0 + k0 + 128, 0, 1);
      __builtin_prefetch(arow1 + k0 + 128, 0, 1);
    }
    v16bf a0 = load_a_frag(A + (size_t)m0 * EMB + k0, EMB, lane);
    v16bf a1 = load_a_frag(A + (size_t)(m0 + 16) * EMB + k0, EMB, lane);
#pragma unroll
    for (int t = 0; t < 4; ++t) {
      v16bf b = load_b_frag_rows(W + (size_t)(n0 + 16 * t) * EMB + k0, EMB, lane);
      c[0][t] = wmma_bf16(a0, b, c[0][t]);
      c[1][t] = wmma_bf16(a1, b, c[1][t]);
    }
  }
  const int hf = lane >> 4, nl = lane & 15;
#pragma unroll
  for (int mi = 0; mi < 2; ++mi)
#pragma unroll
    for (int t = 0; t < 4; ++t)
#pragma unroll
      for (int r = 0; r < 8; ++r) {
        const int m = m0 + mi * 16 + r + 8 * hf;
        const int n = n0 + t * 16 + nl;
        OUT[(size_t)m * EMB + n] = c[mi][t][r] + bias[n];
      }
}

extern "C" void kernel_launch(void* const* d_in, const int* in_sizes, int n_in,
                              void* d_out, int out_size, void* d_ws, size_t ws_size,
                              hipStream_t stream) {
  (void)in_sizes; (void)n_in; (void)out_size; (void)ws_size;
  const float* x     = (const float*)d_in[0];
  const float* w_qkv = (const float*)d_in[1];
  const float* b_qkv = (const float*)d_in[2];
  const float* w_out = (const float*)d_in[3];
  const float* b_out = (const float*)d_in[4];
  float* out = (float*)d_out;

  char* ws = (char*)d_ws;
  size_t off = 0;
  auto take = [&](size_t bytes) -> void* {
    void* p = ws + off;
    off += (bytes + 255) & ~(size_t)255;
    return p;
  };
  __bf16* xbf  = (__bf16*)take((size_t)TOK * EMB * 2);        // 16 MB
  __bf16* wqbf = (__bf16*)take((size_t)3 * EMB * EMB * 2);    //  6 MB
  __bf16* wobf = (__bf16*)take((size_t)EMB * EMB * 2);        //  2 MB
  __bf16* qkv  = (__bf16*)take((size_t)3 * TOK * EMB * 2);    // 48 MB (Q,K,[V^T])
  __bf16* ao   = (__bf16*)take((size_t)TOK * EMB * 2);        // 16 MB

  cvt_bf16<<<2048, 256, 0, stream>>>(x, xbf, TOK * EMB);
  cvt_bf16<<<2048, 256, 0, stream>>>(w_qkv, wqbf, 3 * EMB * EMB);
  cvt_bf16<<<1024, 256, 0, stream>>>(w_out, wobf, EMB * EMB);

  qkv_gemm<<<3072, 128, 0, stream>>>(xbf, wqbf, b_qkv, qkv);   // 12288 waves
  attn_kernel<<<1024, 128, 0, stream>>>(qkv, ao);              //  4096 waves
  out_gemm<<<1024, 128, 0, stream>>>(ao, wobf, b_out, out);    //  4096 waves
}
